// SpatialEncoder_72722386255973
// MI455X (gfx1250) — matrix operand order repaired and checked by reading, not compiled, not run on hardware
//
#include <hip/hip_runtime.h>

typedef __bf16 bf16;
typedef __attribute__((ext_vector_type(16))) __bf16 v16bf;
typedef __attribute__((ext_vector_type(8)))  float  v8f;

#define LN_EPS 1e-5f
#define BN_EPS 1e-5f

// ======================= elementwise / prep kernels =======================

__global__ void k_cvt(const float* __restrict__ in, bf16* __restrict__ out,
                      long long n, int relu) {
  long long i = (long long)blockIdx.x * blockDim.x + threadIdx.x;
  if (i < n) {
    float v = in[i];
    if (relu) v = fmaxf(v, 0.f);
    out[i] = (bf16)v;
  }
}

// in: [R][C] f32 row-major -> out: [C][R] bf16 (conv weight -> B-matrix [K][CO])
__global__ void k_transpose_cvt(const float* __restrict__ in, bf16* __restrict__ out,
                                int R, int C) {
  long long i = (long long)blockIdx.x * blockDim.x + threadIdx.x;
  long long n = (long long)R * C;
  if (i < n) {
    int r = (int)(i / C), c = (int)(i % C);
    out[(long long)c * R + r] = (bf16)in[i];
  }
}

// conv1 weights [64][3][7][7]; input channels identical -> fold: out[k*64+co]=sum_ci w[co][ci][k]
__global__ void k_fold_conv1(const float* __restrict__ w, bf16* __restrict__ out) {
  int i = blockIdx.x * blockDim.x + threadIdx.x;
  if (i < 49 * 64) {
    int k = i >> 6, co = i & 63;
    float s = 0.f;
    for (int ci = 0; ci < 3; ++ci) s += w[co * 147 + ci * 49 + k];
    out[k * 64 + co] = (bf16)s;
  }
}

__global__ void k_bnfold(const float* __restrict__ b, const float* __restrict__ g,
                         const float* __restrict__ m, const float* __restrict__ v,
                         float* __restrict__ scale, float* __restrict__ shift, int C) {
  int i = blockIdx.x * blockDim.x + threadIdx.x;
  if (i < C) {
    float sc = g[i] * rsqrtf(v[i] + BN_EPS);
    scale[i] = sc;
    shift[i] = b[i] - m[i] * sc;
  }
}

// valid[b*10+n] = (sum of mask > 0) ? 1 : 0
__global__ void __launch_bounds__(256) k_valid(const float* __restrict__ om,
                                               float* __restrict__ valid, int HW) {
  __shared__ float red[256];
  int i = blockIdx.x;
  const float* p = om + (long long)i * HW;
  float s = 0.f;
  for (int j = threadIdx.x; j < HW; j += 256) s += p[j];
  red[threadIdx.x] = s; __syncthreads();
  for (int off = 128; off; off >>= 1) {
    if (threadIdx.x < off) red[threadIdx.x] += red[threadIdx.x + off];
    __syncthreads();
  }
  if (threadIdx.x == 0) valid[i] = (red[0] > 0.f) ? 1.f : 0.f;
}

__global__ void k_maxpool(const bf16* __restrict__ in, bf16* __restrict__ out,
                          int NB, int C, int IH, int IW, int OH, int OW) {
  long long i = (long long)blockIdx.x * blockDim.x + threadIdx.x;
  long long n = (long long)NB * C * OH * OW;
  if (i >= n) return;
  int ow = (int)(i % OW); long long t = i / OW;
  int oh = (int)(t % OH); t /= OH;
  int c  = (int)(t % C);
  int nb = (int)(t / C);
  float m = -1e30f;
  for (int kh = 0; kh < 3; ++kh) {
    int ih = oh * 2 - 1 + kh;
    if (ih < 0 || ih >= IH) continue;
    for (int kw = 0; kw < 3; ++kw) {
      int iw = ow * 2 - 1 + kw;
      if (iw < 0 || iw >= IW) continue;
      m = fmaxf(m, (float)in[(((long long)nb * C + c) * IH + ih) * IW + iw]);
    }
  }
  out[i] = (bf16)m;
}

__global__ void k_avgpool(const bf16* __restrict__ in, float* __restrict__ outf,
                          bf16* __restrict__ outb, int NB, int C, int HW) {
  int i = blockIdx.x * blockDim.x + threadIdx.x;
  if (i >= NB * C) return;
  const bf16* p = in + (long long)i * HW;
  float s = 0.f;
  for (int j = 0; j < HW; ++j) s += (float)p[j];
  s /= (float)HW;
  outf[i] = s;
  outb[i] = (bf16)s;
}

__global__ void k_mul(const float* __restrict__ a, const float* __restrict__ b,
                      float* __restrict__ outf, bf16* __restrict__ outb, long long n) {
  long long i = (long long)blockIdx.x * blockDim.x + threadIdx.x;
  if (i < n) {
    float v = a[i] * b[i];
    outf[i] = v;
    outb[i] = (bf16)v;
  }
}

// LayerNorm of (in + resid) over last dim C; optional relu; f32 and bf16 outputs.
// In-place (outf == resid or in) is safe: passes are barrier-separated, writes are per-thread.
__global__ void __launch_bounds__(128) k_ln(const float* __restrict__ in,
                                            const float* __restrict__ resid,
                                            const float* __restrict__ g,
                                            const float* __restrict__ b,
                                            float* __restrict__ outf,
                                            bf16* __restrict__ outb,
                                            int C, int relu) {
  __shared__ float red[128];
  long long row = blockIdx.x;
  const float* x = in + row * C;
  const float* r = resid ? resid + row * C : nullptr;
  float s = 0.f;
  for (int c = threadIdx.x; c < C; c += 128) s += x[c] + (r ? r[c] : 0.f);
  red[threadIdx.x] = s; __syncthreads();
  for (int off = 64; off; off >>= 1) {
    if (threadIdx.x < off) red[threadIdx.x] += red[threadIdx.x + off];
    __syncthreads();
  }
  float mu = red[0] / C; __syncthreads();
  s = 0.f;
  for (int c = threadIdx.x; c < C; c += 128) {
    float v = x[c] + (r ? r[c] : 0.f) - mu;
    s += v * v;
  }
  red[threadIdx.x] = s; __syncthreads();
  for (int off = 64; off; off >>= 1) {
    if (threadIdx.x < off) red[threadIdx.x] += red[threadIdx.x + off];
    __syncthreads();
  }
  float inv = rsqrtf(red[0] / C + LN_EPS); __syncthreads();
  for (int c = threadIdx.x; c < C; c += 128) {
    float v = (x[c] + (r ? r[c] : 0.f) - mu) * inv * g[c] + b[c];
    if (relu) v = fmaxf(v, 0.f);
    if (outf) outf[row * C + c] = v;
    if (outb) outb[row * C + c] = (bf16)v;
  }
}

// Masked MHA over N=10 tokens, D=512, 8 heads of 64. qkv: [80][1536] (q|k|v). out: [80][512]
__global__ void k_attn(const float* __restrict__ qkv, const float* __restrict__ valid,
                       float* __restrict__ out) {
  int b = blockIdx.x, h = blockIdx.y;
  int n = threadIdx.x;
  if (n >= 10) return;
  const float* base = qkv + (long long)b * 10 * 1536;
  float q[64];
  for (int d = 0; d < 64; ++d) q[d] = base[n * 1536 + h * 64 + d];
  float s[10]; float mx = -1e30f;
  for (int j = 0; j < 10; ++j) {
    const float* kr = base + j * 1536 + 512 + h * 64;
    float acc = 0.f;
    for (int d = 0; d < 64; ++d) acc += q[d] * kr[d];
    acc *= 0.125f;                       // 1/sqrt(64)
    if (valid[b * 10 + j] == 0.f) acc = -1e9f;
    s[j] = acc; mx = fmaxf(mx, acc);
  }
  float den = 0.f;
  for (int j = 0; j < 10; ++j) { s[j] = expf(s[j] - mx); den += s[j]; }
  float inv = 1.f / den;
  float* orow = out + (long long)(b * 10 + n) * 512 + h * 64;
  for (int d = 0; d < 64; ++d) {
    float acc = 0.f;
    for (int j = 0; j < 10; ++j) acc += s[j] * base[j * 1536 + 1024 + h * 64 + d];
    orow[d] = acc * inv;
  }
}

__global__ void k_final(const float* __restrict__ logits, const float* __restrict__ valid,
                        float* __restrict__ out) {
  int i = threadIdx.x + blockIdx.x * blockDim.x;
  if (i < 80) {
    float v = valid[i];
    out[i]      = (v > 0.f) ? logits[i] : -1e9f;
    out[80 + i] = v;
  }
}

// ======================= WMMA GEMM / implicit-GEMM conv =======================
// 128 threads = 4 wave32. Block tile: 32(M) x 64(N). Wave (wr,wc) owns rows wr*16..+15,
// cols wc*32..+31 via two 16x16 WMMAs sharing one A fragment (2x arithmetic intensity).
// Fragment layouts per CDNA5 ISA 7.12.2:
//   A 16x32 bf16: lane<16 -> M=lane, K 0-7 & 16-23 ; lane>=16 -> M=lane-16, K 8-15 & 24-31
//   B 32x16 bf16 (staged transposed in LDS): lane<16 -> K 0-7 & 16-23 ; lane>=16 -> K 8-15 & 24-31
//   C f32:       element i -> row i (lanes 0-15) / row i+8 (lanes 16-31), col = lane&15
// A panel staging uses CDNA5 async copy (global_load_async_to_lds_b128 + ASYNCcnt).

__device__ __forceinline__ void async_a_b128(bf16* lds_dst, const bf16* gsrc) {
  unsigned lds = (unsigned)(unsigned long long)(void*)lds_dst;   // low 32b of generic ptr = LDS offset
  unsigned long long ga = (unsigned long long)(const void*)gsrc;
  asm volatile("global_load_async_to_lds_b128 %0, %1, off" :: "v"(lds), "v"(ga) : "memory");
}
__device__ __forceinline__ void wait_async() {
  asm volatile("s_wait_asynccnt 0x0" ::: "memory");
}

__global__ void __launch_bounds__(128)
k_gemm_wmma(const bf16* __restrict__ A, const bf16* __restrict__ B,
            const float* __restrict__ bias, float* __restrict__ C,
            int M, int N, int K) {
  __shared__ bf16 As[32 * 32];
  __shared__ bf16 Bs[64 * 32];   // transposed: Bs[n][k]
  const int tid = threadIdx.x;
  const int lane = tid & 31, wave = tid >> 5;
  const int wr = wave >> 1, wc = wave & 1;
  const int m0 = blockIdx.x * 32, n0 = blockIdx.y * 64;
  const int r = tid >> 2;            // 0..31
  const int cg = (tid & 3) << 3;     // 0,8,16,24
  v8f acc[2] = {{}, {}};
  for (int kt = 0; kt < K; kt += 32) {
    if (kt + 32 < K) __builtin_prefetch(&B[(long long)(kt + 32) * N], 0, 1);
    // ---- A tile 32x32: async direct-to-LDS fast path ----
    {
      int gm = m0 + r, gk = kt + cg;
      bf16* dst = &As[r * 32 + cg];
      if (gm < M && gk + 8 <= K) {
        async_a_b128(dst, &A[(long long)gm * K + gk]);
      } else {
#pragma unroll
        for (int j = 0; j < 8; ++j)
          dst[j] = (gm < M && gk + j < K) ? A[(long long)gm * K + gk + j] : (bf16)0.f;
      }
    }
    // ---- B tile 32x64 -> Bs[n][k] (vector global load, transposed LDS scatter) ----
#pragma unroll
    for (int h = 0; h < 2; ++h) {
      int gk2 = kt + r;
      int nloc = cg + h * 32;
      int gn = n0 + nloc;
      bf16 tb[8];
      if (gk2 < K && gn + 8 <= N) {
        *(uint4*)tb = *(const uint4*)&B[(long long)gk2 * N + gn];
      } else {
#pragma unroll
        for (int j = 0; j < 8; ++j)
          tb[j] = (gk2 < K && gn + j < N) ? B[(long long)gk2 * N + gn + j] : (bf16)0.f;
      }
#pragma unroll
      for (int j = 0; j < 8; ++j) Bs[(nloc + j) * 32 + r] = tb[j];
    }
    wait_async();
    __syncthreads();
    v16bf a, b0v, b1v;
    {
      int mrow = wr * 16 + (lane & 15);
      int kb = (lane & 16) ? 8 : 0;
      const bf16* pa = &As[mrow * 32 + kb];
#pragma unroll
      for (int t = 0; t < 8; ++t) { a[t] = pa[t]; a[8 + t] = pa[16 + t]; }
      int kb2 = (lane & 16) ? 8 : 0;
      const bf16* pb0 = &Bs[(wc * 32 + (lane & 15)) * 32 + kb2];
      const bf16* pb1 = &Bs[(wc * 32 + 16 + (lane & 15)) * 32 + kb2];
#pragma unroll
      for (int t = 0; t < 8; ++t) { b0v[t] = pb0[t]; b0v[8 + t] = pb0[16 + t]; b1v[t] = pb1[t]; b1v[8 + t] = pb1[16 + t]; }
    }
    acc[0] = __builtin_amdgcn_wmma_f32_16x16x32_bf16(false, a, false, b0v, (short)0,
                                                     acc[0], false, false);
    acc[1] = __builtin_amdgcn_wmma_f32_16x16x32_bf16(false, a, false, b1v, (short)0,
                                                     acc[1], false, false);
    __syncthreads();
  }
#pragma unroll
  for (int h = 0; h < 2; ++h) {
    int col = n0 + wc * 32 + h * 16 + (lane & 15);
    if (col < N) {
      float bs = bias ? bias[col] : 0.f;
#pragma unroll
      for (int i = 0; i < 8; ++i) {
        int row = m0 + wr * 16 + i + ((lane & 16) ? 8 : 0);
        if (row < M) C[(long long)row * N + col] = acc[h][i] + bs;
      }
    }
  }
}

// Implicit-GEMM conv + folded BN + optional residual + optional ReLU, bf16 out.
// in: [NB][CI][IH][IW] bf16, Bw: [K=CI*KH*KW][CO] bf16, out: [NB][CO][OH][OW] bf16
__global__ void __launch_bounds__(128)
k_conv_wmma(const bf16* __restrict__ in, const bf16* __restrict__ Bw,
            const float* __restrict__ scale, const float* __restrict__ shift,
            const bf16* __restrict__ resid, bf16* __restrict__ out,
            int NB, int CI, int IH, int IW, int CO, int OH, int OW,
            int KH, int KW, int stride, int pad, int relu) {
  __shared__ bf16 As[32 * 32];
  __shared__ bf16 Bs[64 * 32];   // transposed: Bs[n][k]
  const int tid = threadIdx.x;
  const int lane = tid & 31, wave = tid >> 5;
  const int wr = wave >> 1, wc = wave & 1;
  const int OHW = OH * OW, KHW = KH * KW;
  const int M = NB * OHW;
  const int K = CI * KHW;
  const int m0 = blockIdx.x * 32, n0 = blockIdx.y * 64;
  const int r = tid >> 2;
  const int cg = (tid & 3) << 3;
  v8f acc[2] = {{}, {}};
  for (int kt = 0; kt < K; kt += 32) {
    if (kt + 32 < K) __builtin_prefetch(&Bw[(long long)(kt + 32) * CO], 0, 1);
    // ---- A tile: im2col gather (8 elements / thread) ----
    {
      int gm = m0 + r;
      bf16* dst = &As[r * 32 + cg];
      int nb = 0, oh = 0, ow = 0;
      bool mok = gm < M;
      if (mok) {
        nb = gm / OHW;
        int rem = gm - nb * OHW;
        oh = rem / OW; ow = rem - oh * OW;
      }
#pragma unroll
      for (int j = 0; j < 8; ++j) {
        int gk = kt + cg + j;
        bf16 val = (bf16)0.f;
        if (mok && gk < K) {
          int ci = gk / KHW;
          int kr = gk - ci * KHW;
          int kh = kr / KW, kw = kr - kh * KW;
          int ih = oh * stride - pad + kh;
          int iw = ow * stride - pad + kw;
          if (ih >= 0 && ih < IH && iw >= 0 && iw < IW)
            val = in[(((long long)nb * CI + ci) * IH + ih) * IW + iw];
        }
        dst[j] = val;
      }
    }
    // ---- B tile 32x64 -> Bs[n][k] ----
#pragma unroll
    for (int h = 0; h < 2; ++h) {
      int gk2 = kt + r;
      int nloc = cg + h * 32;
      int gn = n0 + nloc;
      bf16 tb[8];
      if (gk2 < K && gn + 8 <= CO) {
        *(uint4*)tb = *(const uint4*)&Bw[(long long)gk2 * CO + gn];
      } else {
#pragma unroll
        for (int j = 0; j < 8; ++j)
          tb[j] = (gk2 < K && gn + j < CO) ? Bw[(long long)gk2 * CO + gn + j] : (bf16)0.f;
      }
#pragma unroll
      for (int j = 0; j < 8; ++j) Bs[(nloc + j) * 32 + r] = tb[j];
    }
    __syncthreads();
    v16bf a, b0v, b1v;
    {
      int mrow = wr * 16 + (lane & 15);
      int kb = (lane & 16) ? 8 : 0;
      const bf16* pa = &As[mrow * 32 + kb];
#pragma unroll
      for (int t = 0; t < 8; ++t) { a[t] = pa[t]; a[8 + t] = pa[16 + t]; }
      int kb2 = (lane & 16) ? 8 : 0;
      const bf16* pb0 = &Bs[(wc * 32 + (lane & 15)) * 32 + kb2];
      const bf16* pb1 = &Bs[(wc * 32 + 16 + (lane & 15)) * 32 + kb2];
#pragma unroll
      for (int t = 0; t < 8; ++t) { b0v[t] = pb0[t]; b0v[8 + t] = pb0[16 + t]; b1v[t] = pb1[t]; b1v[8 + t] = pb1[16 + t]; }
    }
    acc[0] = __builtin_amdgcn_wmma_f32_16x16x32_bf16(false, a, false, b0v, (short)0,
                                                     acc[0], false, false);
    acc[1] = __builtin_amdgcn_wmma_f32_16x16x32_bf16(false, a, false, b1v, (short)0,
                                                     acc[1], false, false);
    __syncthreads();
  }
#pragma unroll
  for (int h = 0; h < 2; ++h) {
    int col = n0 + wc * 32 + h * 16 + (lane & 15);
    if (col < CO) {
      float sc = scale[col], sh = shift[col];
#pragma unroll
      for (int i = 0; i < 8; ++i) {
        int row = m0 + wr * 16 + i + ((lane & 16) ? 8 : 0);
        if (row < M) {
          int nb  = row / OHW;
          int rem = row - nb * OHW;
          int oh = rem / OW, ow = rem - oh * OW;
          long long oidx = (((long long)nb * CO + col) * OH + oh) * OW + ow;
          float v = acc[h][i] * sc + sh;
          if (resid) v += (float)resid[oidx];
          if (relu) v = fmaxf(v, 0.f);
          out[oidx] = (bf16)v;
        }
      }
    }
  }
}

// ============================== host driver ==============================

struct LinP { const float *b, *w; };
struct LNP  { const float *b, *g; };
struct EncP { LinP ff1, ff2; LNP ln1, ln2; LinP out, qkv; };
struct BlkP { const float* bn1[4]; const float* bn2[4]; const float *c1, *c2;
              bool ds; const float* dsbn[4]; const float* dsc; };

extern "C" void kernel_launch(void* const* d_in, const int* in_sizes, int n_in,
                              void* d_out, int out_size, void* d_ws, size_t ws_size,
                              hipStream_t stream) {
  (void)in_sizes; (void)n_in; (void)out_size; (void)ws_size;
  const float* target = (const float*)d_in[0];   // [8][1][224][224]
  const float* objm   = (const float*)d_in[1];   // [8][10][1][224][224]
  // d_in[2] = bboxes: only feeds _edge_feats which is dead in the reference.

  // ---- parse params pytree (JAX dict order: enc, obj, outp, resnet, tgt) ----
  int pi = 3;
  auto P = [&]() { return (const float*)d_in[pi++]; };
  EncP enc[6];
  for (int l = 0; l < 6; ++l) {
    enc[l].ff1.b = P(); enc[l].ff1.w = P();
    enc[l].ff2.b = P(); enc[l].ff2.w = P();
    enc[l].ln1.b = P(); enc[l].ln1.g = P();
    enc[l].ln2.b = P(); enc[l].ln2.g = P();
    enc[l].out.b = P(); enc[l].out.w = P();
    enc[l].qkv.b = P(); enc[l].qkv.w = P();
  }
  LinP obj_l1, obj_l2; LNP obj_ln;
  obj_l1.b = P(); obj_l1.w = P(); obj_l2.b = P(); obj_l2.w = P();
  obj_ln.b = P(); obj_ln.g = P();
  LinP op_l1, op_l2, op_l3; LNP op_ln1, op_ln2;
  op_l1.b = P(); op_l1.w = P(); op_l2.b = P(); op_l2.w = P();
  op_l3.b = P(); op_l3.w = P();
  op_ln1.b = P(); op_ln1.g = P(); op_ln2.b = P(); op_ln2.g = P();
  const float* rbn1[4]; for (int j = 0; j < 4; ++j) rbn1[j] = P();
  const float* rconv1 = P();
  LinP fc; fc.b = P(); fc.w = P();
  BlkP blk[4][2];
  for (int st = 0; st < 4; ++st)
    for (int bi = 0; bi < 2; ++bi) {
      BlkP& B = blk[st][bi];
      for (int j = 0; j < 4; ++j) B.bn1[j] = P();
      for (int j = 0; j < 4; ++j) B.bn2[j] = P();
      B.c1 = P(); B.c2 = P();
      B.ds = (bi == 0 && st > 0);
      if (B.ds) { for (int j = 0; j < 4; ++j) B.dsbn[j] = P(); B.dsc = P(); }
    }
  LinP tg_l1, tg_l2; LNP tg_ln;
  tg_l1.b = P(); tg_l1.w = P(); tg_l2.b = P(); tg_l2.w = P();
  tg_ln.b = P(); tg_ln.g = P();

  // ---- workspace bump allocator ----
  char* wsb = (char*)d_ws; size_t cur = 0;
  auto alloc = [&](size_t bytes) -> void* {
    void* p = wsb + cur; cur += (bytes + 255) & ~(size_t)255; return p;
  };

  // ---- launch helpers ----
  auto cvt = [&](const float* in, bf16* out, long long n, int relu) {
    k_cvt<<<(unsigned)((n + 255) / 256), 256, 0, stream>>>(in, out, n, relu);
  };
  auto trans_cvt = [&](const float* in, int R, int C) -> bf16* {
    bf16* o = (bf16*)alloc((size_t)R * C * 2);
    long long n = (long long)R * C;
    k_transpose_cvt<<<(unsigned)((n + 255) / 256), 256, 0, stream>>>(in, o, R, C);
    return o;
  };
  auto cvt_w = [&](const float* in, long long n) -> bf16* {
    bf16* o = (bf16*)alloc((size_t)n * 2);
    cvt(in, o, n, 0);
    return o;
  };
  auto bnfold = [&](const float* const bn[4], int C, const float** s, const float** h) {
    float* sc = (float*)alloc(C * 4); float* sh = (float*)alloc(C * 4);
    k_bnfold<<<(C + 255) / 256, 256, 0, stream>>>(bn[0], bn[1], bn[2], bn[3], sc, sh, C);
    *s = sc; *h = sh;
  };
  auto conv = [&](const bf16* in, const bf16* Bw, const float* sc, const float* sh,
                  const bf16* resid, bf16* out, int NB, int CI, int IH, int IW, int CO,
                  int KH, int KW, int stride, int pad, int relu, int OH, int OW) {
    int M = NB * OH * OW;
    dim3 g((M + 31) / 32, (CO + 63) / 64);
    k_conv_wmma<<<g, 128, 0, stream>>>(in, Bw, sc, sh, resid, out, NB, CI, IH, IW, CO,
                                       OH, OW, KH, KW, stride, pad, relu);
  };
  auto gemm = [&](const bf16* A, const bf16* B, const float* bias, float* C,
                  int M, int N, int K) {
    dim3 g((M + 31) / 32, (N + 63) / 64);
    k_gemm_wmma<<<g, 128, 0, stream>>>(A, B, bias, C, M, N, K);
  };
  auto lnrun = [&](const float* in, const float* resid, const float* g, const float* b,
                   float* outf, bf16* outb, int rows, int C, int relu) {
    k_ln<<<rows, 128, 0, stream>>>(in, resid, g, b, outf, outb, C, relu);
  };

  // ---- validity flags + input conversion ----
  float* validf = (float*)alloc(80 * 4);
  k_valid<<<80, 256, 0, stream>>>(objm, validf, 224 * 224);
  bf16* inb = (bf16*)alloc(88ll * 50176 * 2);        // [88][1][224][224]
  cvt(target, inb, 8ll * 50176, 0);
  cvt(objm, inb + 8ll * 50176, 80ll * 50176, 0);

  // ---- ResNet-18 over the 88-image batch ----
  bf16* w1b = (bf16*)alloc(49 * 64 * 2);
  k_fold_conv1<<<(49 * 64 + 255) / 256, 256, 0, stream>>>(rconv1, w1b);
  const float *s1, *h1; bnfold(rbn1, 64, &s1, &h1);
  bf16* act1 = (bf16*)alloc(88ll * 64 * 112 * 112 * 2);
  conv(inb, w1b, s1, h1, nullptr, act1, 88, 1, 224, 224, 64, 7, 7, 2, 3, 1, 112, 112);

  size_t bsz = 88ll * 64 * 56 * 56 * 2;              // largest post-pool activation
  bf16* b0 = (bf16*)alloc(bsz);
  bf16* b1 = (bf16*)alloc(bsz);
  bf16* b2 = (bf16*)alloc(bsz);
  {
    long long n = 88ll * 64 * 56 * 56;
    k_maxpool<<<(unsigned)((n + 255) / 256), 256, 0, stream>>>(act1, b0, 88, 64, 112, 112,
                                                               56, 56);
  }
  bf16 *X = b0, *T = b1, *Z = b2;
  int cin = 64, H = 56;
  const int couts[4] = {64, 128, 256, 512};
  const int strides[4] = {1, 2, 2, 2};
  for (int st = 0; st < 4; ++st) {
    int cout = couts[st];
    for (int bi = 0; bi < 2; ++bi) {
      BlkP& B = blk[st][bi];
      int s = (bi == 0) ? strides[st] : 1;
      int OH = H / s;
      const float *sc1, *sh1, *sc2, *sh2;
      bnfold(B.bn1, cout, &sc1, &sh1);
      bnfold(B.bn2, cout, &sc2, &sh2);
      bf16* c1b = trans_cvt(B.c1, cout, cin * 9);
      bf16* c2b = trans_cvt(B.c2, cout, cout * 9);
      conv(X, c1b, sc1, sh1, nullptr, T, 88, cin, H, H, cout, 3, 3, s, 1, 1, OH, OH);
      const bf16* resid;
      if (B.ds) {
        const float *scd, *shd; bnfold(B.dsbn, cout, &scd, &shd);
        bf16* dsb = trans_cvt(B.dsc, cout, cin);
        conv(X, dsb, scd, shd, nullptr, Z, 88, cin, H, H, cout, 1, 1, s, 0, 0, OH, OH);
        resid = Z;   // in-place residual read in epilogue is element-wise safe
      } else {
        resid = X;
      }
      conv(T, c2b, sc2, sh2, resid, Z, 88, cout, OH, OH, cout, 3, 3, 1, 1, 1, OH, OH);
      bf16* t = X; X = Z; Z = t;   // new input = Z; old X free
      cin = cout; H = OH;
    }
  }
  // X = [88][512][7][7]
  float* feat  = (float*)alloc(88 * 512 * 4);
  bf16*  featb = (bf16*)alloc(88 * 512 * 2);
  k_avgpool<<<(88 * 512 + 255) / 256, 256, 0, stream>>>(X, feat, featb, 88, 512, 49);
  bf16* fcwb = cvt_w(fc.w, 512ll * 1024);
  float* feats = (float*)alloc(88 * 1024 * 4);
  gemm(featb, fcwb, fc.b, feats, 88, 1024, 512);
  bf16* featsb = (bf16*)alloc(88 * 1024 * 2);
  cvt(feats, featsb, 88 * 1024, 0);

  // ---- target / object embedding heads ----
  bf16* tl1w = cvt_w(tg_l1.w, 1024ll * 1024);
  float* t1 = (float*)alloc(8 * 1024 * 4);
  gemm(featsb, tl1w, tg_l1.b, t1, 8, 1024, 1024);
  bf16* t2b = (bf16*)alloc(8 * 1024 * 2);
  lnrun(t1, nullptr, tg_ln.g, tg_ln.b, nullptr, t2b, 8, 1024, 1);
  bf16* tl2w = cvt_w(tg_l2.w, 1024ll * 5120);
  float* t3 = (float*)alloc(8 * 5120 * 4);
  gemm(t2b, tl2w, tg_l2.b, t3, 8, 5120, 1024);

  bf16* ol1w = cvt_w(obj_l1.w, 10240ll * 1024);
  float* o1 = (float*)alloc(8 * 1024 * 4);
  gemm(featsb + 8 * 1024, ol1w, obj_l1.b, o1, 8, 1024, 10240);  // rows 8..88 = [8][10240]
  bf16* o2b = (bf16*)alloc(8 * 1024 * 2);
  lnrun(o1, nullptr, obj_ln.g, obj_ln.b, nullptr, o2b, 8, 1024, 1);
  bf16* ol2w = cvt_w(obj_l2.w, 1024ll * 5120);
  float* o3 = (float*)alloc(8 * 5120 * 4);
  gemm(o2b, ol2w, obj_l2.b, o3, 8, 5120, 1024);

  float* x  = (float*)alloc(80 * 512 * 4);
  bf16*  xb = (bf16*)alloc(80 * 512 * 2);
  k_mul<<<(40960 + 255) / 256, 256, 0, stream>>>(o3, t3, x, xb, 40960);

  // ---- transformer encoder ----
  bf16 *qkvw[6], *outw[6], *ff1w[6], *ff2w[6];
  for (int l = 0; l < 6; ++l) {
    qkvw[l] = cvt_w(enc[l].qkv.w, 512ll * 1536);
    outw[l] = cvt_w(enc[l].out.w, 512ll * 512);
    ff1w[l] = cvt_w(enc[l].ff1.w, 512ll * 2048);
    ff2w[l] = cvt_w(enc[l].ff2.w, 2048ll * 512);
  }
  float* qkv   = (float*)alloc(80 * 1536 * 4);
  float* attno = (float*)alloc(80 * 512 * 4);
  bf16*  attnb = (bf16*)alloc(80 * 512 * 2);
  float* proj  = (float*)alloc(80 * 512 * 4);
  float* ff1   = (float*)alloc(80 * 2048 * 4);
  bf16*  ff1b  = (bf16*)alloc(80 * 2048 * 2);
  float* ff2   = (float*)alloc(80 * 512 * 4);
  for (int l = 0; l < 6; ++l) {
    gemm(xb, qkvw[l], enc[l].qkv.b, qkv, 80, 1536, 512);
    k_attn<<<dim3(8, 8), 16, 0, stream>>>(qkv, validf, attno);
    cvt(attno, attnb, 80 * 512, 0);
    gemm(attnb, outw[l], enc[l].out.b, proj, 80, 512, 512);
    lnrun(proj, x, enc[l].ln1.g, enc[l].ln1.b, x, xb, 80, 512, 0);
    gemm(xb, ff1w[l], enc[l].ff1.b, ff1, 80, 2048, 512);
    cvt(ff1, ff1b, 80 * 2048, 1);                       // fused ReLU
    gemm(ff1b, ff2w[l], enc[l].ff2.b, ff2, 80, 512, 2048);
    lnrun(ff2, x, enc[l].ln2.g, enc[l].ln2.b, x, xb, 80, 512, 0);
  }

  // ---- output head ----
  bf16* pl1w = cvt_w(op_l1.w, 512ll * 1024);
  float* y1 = (float*)alloc(80 * 1024 * 4);
  gemm(xb, pl1w, op_l1.b, y1, 80, 1024, 512);
  bf16* y1b = (bf16*)alloc(80 * 1024 * 2);
  lnrun(y1, nullptr, op_ln1.g, op_ln1.b, nullptr, y1b, 80, 1024, 1);
  bf16* pl2w = cvt_w(op_l2.w, 1024ll * 512);
  float* y2 = (float*)alloc(80 * 512 * 4);
  gemm(y1b, pl2w, op_l2.b, y2, 80, 512, 1024);
  bf16* y2b = (bf16*)alloc(80 * 512 * 2);
  lnrun(y2, nullptr, op_ln2.g, op_ln2.b, nullptr, y2b, 80, 512, 1);
  bf16* pl3w = cvt_w(op_l3.w, 512);
  float* logits = (float*)alloc(80 * 4);
  gemm(y2b, pl3w, op_l3.b, logits, 80, 1, 512);
  k_final<<<1, 128, 0, stream>>>(logits, validf, (float*)d_out);
}
